// SMKMoE_33097017983636
// MI455X (gfx1250) — matrix operand order repaired and checked by reading, not compiled, not run on hardware
//
#include <hip/hip_runtime.h>
#include <math.h>

#define N_TOK 4096
#define C_DIM 1024
#define E_NUM 16
#define I_DIM 256

typedef __attribute__((ext_vector_type(16))) __bf16 v16bf;
typedef __attribute__((ext_vector_type(8)))  __bf16 v8bf;
typedef __attribute__((ext_vector_type(8)))  float  v8f;
typedef __attribute__((ext_vector_type(4)))  float  v4f;

// ---- WMMA fragment loaders (bf16, 16x16x32) ----
// A (16x32, MxK), row-major source with leading dim ld.
// Layout (matches B): lanes 0-15 = rows, K(0..15) in V0..7;
// lanes 16-31 same rows, K(16..31).
__device__ __forceinline__ v16bf load_a_frag(const __bf16* src, int ld, int k0) {
  int lane = threadIdx.x & 31;
  int m = lane & 15;
  int kh = (lane >> 4) * 16;
  const __bf16* p = src + (size_t)m * ld + k0 + kh;
  v8bf lo = *(const v8bf*)p;          // K = k0+kh .. +7
  v8bf hi = *(const v8bf*)(p + 8);    // K = k0+kh+8 .. +15
  v16bf r;
#pragma unroll
  for (int i = 0; i < 8; ++i) { r[i] = lo[i]; r[i + 8] = hi[i]; }
  return r;
}

// B (32x16, KxN) where B[k][n] = w[(n0+n)*ld + k0 + k] (w row-major along K).
// Layout used here (not the ISA table order): lanes 0-15 = cols with K=0..15 (2/VGPR), lanes 16-31 = K=16..31.
// -> one contiguous 32B load per lane.
__device__ __forceinline__ v16bf load_b_frag(const __bf16* w, int ld, int n0, int k0) {
  int lane = threadIdx.x & 31;
  int n = n0 + (lane & 15);
  int kb = (lane >> 4) * 16;
  return *(const v16bf*)(w + (size_t)n * ld + k0 + kb);
}

// ---- sim column norms ----
__global__ __launch_bounds__(256) void snorm_kernel(const float* __restrict__ sim,
                                                    float* __restrict__ snorm) {
  __shared__ float sm[16][16];
  int e = threadIdx.x & 15, g = threadIdx.x >> 4;
  float s = 0.0f;
  for (int c = g; c < C_DIM; c += 16) { float v = sim[(size_t)c * E_NUM + e]; s += v * v; }
  sm[g][e] = s;
  __syncthreads();
  if (threadIdx.x < 16) {
    float t = 0.0f;
#pragma unroll
    for (int g2 = 0; g2 < 16; ++g2) t += sm[g2][threadIdx.x];
    snorm[threadIdx.x] = fmaxf(sqrtf(t), 1e-12f);
  }
}

// ---- scores + mask count + x->bf16 cast ----
__global__ __launch_bounds__(256) void score_kernel(
    const float* __restrict__ x, const float* __restrict__ sim,
    const float* __restrict__ thr, const float* __restrict__ snorm,
    float* __restrict__ scores, float* __restrict__ kout,
    __bf16* __restrict__ xb) {
  const int n = blockIdx.x;
  const int t = threadIdx.x;
  const float* xr = x + (size_t)n * C_DIM;
  float vals[17];
#pragma unroll
  for (int v = 0; v < 17; ++v) vals[v] = 0.0f;
  const int c0 = t * 4;
  v4f xv = *(const v4f*)(xr + c0);
#pragma unroll
  for (int j = 0; j < 4; ++j) {
    float xvj = xv[j];
    vals[0] += xvj * xvj;
    const float* srow = sim + (size_t)(c0 + j) * E_NUM;
#pragma unroll
    for (int ee = 0; ee < 16; ++ee) vals[1 + ee] += xvj * srow[ee];
    xb[(size_t)n * C_DIM + c0 + j] = (__bf16)xvj;
  }
#pragma unroll
  for (int off = 16; off > 0; off >>= 1)
#pragma unroll
    for (int v = 0; v < 17; ++v) vals[v] += __shfl_xor(vals[v], off, 32);
  __shared__ float red[8][17];
  const int wid = t >> 5, lane = t & 31;
  if (lane == 0)
#pragma unroll
    for (int v = 0; v < 17; ++v) red[wid][v] = vals[v];
  __syncthreads();
  if (t < 16) {
    float ss = 0.0f, d = 0.0f;
#pragma unroll
    for (int w = 0; w < 8; ++w) { ss += red[w][0]; d += red[w][1 + t]; }
    float sc = d / (fmaxf(sqrtf(ss), 1e-12f) * snorm[t]);
    scores[(size_t)n * E_NUM + t] = sc;
    unsigned long long b = __ballot(sc > thr[0]);
    if (t == 0) kout[n] = (float)__popcll(b & 0xFFFFull);
  }
}

// ---- W1/W2 -> bf16 ----
__global__ __launch_bounds__(256) void cvtw_kernel(const float* __restrict__ W1,
                                                   const float* __restrict__ W2,
                                                   __bf16* __restrict__ w1b,
                                                   __bf16* __restrict__ w2b) {
  size_t i = (size_t)blockIdx.x * blockDim.x + threadIdx.x;
  const size_t total = (size_t)E_NUM * I_DIM * C_DIM;
  if (i < total) { w1b[i] = (__bf16)W1[i]; w2b[i] = (__bf16)W2[i]; }
}

// ---- fused per-(token-tile, expert) FFN: gemm1 -> gelu -> gemm2 -> masked store ----
__global__ __launch_bounds__(256) void moe_kernel(
    const __bf16* __restrict__ xb, const __bf16* __restrict__ w1b,
    const __bf16* __restrict__ w2b, const float* __restrict__ scores,
    const float* __restrict__ thr, float* __restrict__ eo) {
  __shared__ __bf16 hS[16][264];   // 16 tokens x 256 intermediate, +8 pad (bank spread)
  __shared__ float maskS[16];
  const int tIdx = blockIdx.x;     // token tile 0..255
  const int e = blockIdx.y;        // expert
  const int wid = threadIdx.x >> 5;
  const int lane = threadIdx.x & 31;
  const int row0_tok = tIdx * 16;

  if (threadIdx.x < 16) {
    float s = scores[(size_t)(row0_tok + threadIdx.x) * E_NUM + e];
    maskS[threadIdx.x] = (s > thr[0]) ? 1.0f : 0.0f;
  }

  // Phase A: h = gelu(x @ W1[e]^T), each wave does 2 of 16 I-tiles
  const __bf16* xrow = xb + (size_t)row0_tok * C_DIM;
  const __bf16* w1e = w1b + (size_t)e * I_DIM * C_DIM;
  const int i0 = wid * 16;
  const int i1 = (wid + 8) * 16;
  v8f acc0 = {}; v8f acc1 = {};
  for (int k0 = 0; k0 < C_DIM; k0 += 32) {
    v16bf a  = load_a_frag(xrow, C_DIM, k0);
    v16bf b0 = load_b_frag(w1e, C_DIM, i0, k0);
    v16bf b1 = load_b_frag(w1e, C_DIM, i1, k0);
    acc0 = __builtin_amdgcn_wmma_f32_16x16x32_bf16(false, a, false, b0, (short)0, acc0, false, false);
    acc1 = __builtin_amdgcn_wmma_f32_16x16x32_bf16(false, a, false, b1, (short)0, acc1, false, false);
  }
  const int mbase = (lane >> 4) * 8;   // C/D layout: elem r -> row mbase+r, col lane&15
  const int ncol = lane & 15;
#pragma unroll
  for (int r = 0; r < 8; ++r) {
    float g0 = acc0[r]; g0 = 0.5f * g0 * (1.0f + erff(g0 * 0.70710678f));
    float g1 = acc1[r]; g1 = 0.5f * g1 * (1.0f + erff(g1 * 0.70710678f));
    hS[mbase + r][i0 + ncol] = (__bf16)g0;
    hS[mbase + r][i1 + ncol] = (__bf16)g1;
  }
  __syncthreads();

  // Phase B: out = h @ W2[e]^T, each wave does 8 of 64 C-tiles
  const __bf16* w2e = w2b + (size_t)e * C_DIM * I_DIM;
  const __bf16* hP = &hS[0][0];
  v8f acc[8] = {};
  for (int k0 = 0; k0 < I_DIM; k0 += 32) {
    v16bf a = load_a_frag(hP, 264, k0);
#pragma unroll
    for (int j = 0; j < 8; ++j) {
      int cb = (wid + 8 * j) * 16;
      v16bf b = load_b_frag(w2e, I_DIM, cb, k0);
      acc[j] = __builtin_amdgcn_wmma_f32_16x16x32_bf16(false, a, false, b, (short)0, acc[j], false, false);
    }
  }
  float mk[8];
#pragma unroll
  for (int r = 0; r < 8; ++r) mk[r] = maskS[mbase + r];
#pragma unroll
  for (int j = 0; j < 8; ++j) {
    int col = (wid + 8 * j) * 16 + ncol;
#pragma unroll
    for (int r = 0; r < 8; ++r) {
      size_t tok = (size_t)(row0_tok + mbase + r);
      eo[tok * (E_NUM * C_DIM) + (size_t)e * C_DIM + col] = acc[j][r] * mk[r];
    }
  }
}

// ---- final = sum over experts (mask already applied once) ----
__global__ __launch_bounds__(256) void final_kernel(const float* __restrict__ eo,
                                                    float* __restrict__ fin) {
  size_t i = (size_t)blockIdx.x * blockDim.x + threadIdx.x;
  if (i >= (size_t)N_TOK * C_DIM) return;
  size_t n = i >> 10, c = i & 1023;
  const float* p = eo + n * (size_t)(E_NUM * C_DIM) + c;
  float s = 0.0f;
#pragma unroll
  for (int ee = 0; ee < 16; ++ee) s += p[(size_t)ee * C_DIM];
  fin[i] = s;
}

extern "C" void kernel_launch(void* const* d_in, const int* in_sizes, int n_in,
                              void* d_out, int out_size, void* d_ws, size_t ws_size,
                              hipStream_t stream) {
  (void)in_sizes; (void)n_in; (void)out_size; (void)ws_size;
  const float* x   = (const float*)d_in[0];
  const float* sim = (const float*)d_in[1];
  const float* thr = (const float*)d_in[2];
  const float* W1  = (const float*)d_in[3];
  const float* W2  = (const float*)d_in[4];

  float* out    = (float*)d_out;
  float* fin    = out;                                       // [4096,1024]
  float* scores = out + (size_t)N_TOK * C_DIM;               // [4096,16]
  float* eo     = scores + (size_t)N_TOK * E_NUM;            // [4096,16,1024]
  float* kout   = eo + (size_t)N_TOK * E_NUM * C_DIM;        // [4096]

  __bf16* xb  = (__bf16*)d_ws;                               // 8 MB
  __bf16* w1b = xb + (size_t)N_TOK * C_DIM;                  // 8 MB
  __bf16* w2b = w1b + (size_t)E_NUM * I_DIM * C_DIM;         // 8 MB
  float* snorm = (float*)(w2b + (size_t)E_NUM * C_DIM * I_DIM);

  hipLaunchKernelGGL(snorm_kernel, dim3(1), dim3(256), 0, stream, sim, snorm);
  hipLaunchKernelGGL(score_kernel, dim3(N_TOK), dim3(256), 0, stream,
                     x, sim, thr, snorm, scores, kout, xb);
  hipLaunchKernelGGL(cvtw_kernel, dim3((E_NUM * I_DIM * C_DIM) / 256), dim3(256), 0, stream,
                     W1, W2, w1b, w2b);
  hipLaunchKernelGGL(moe_kernel, dim3(N_TOK / 16, E_NUM), dim3(256), 0, stream,
                     xb, w1b, w2b, scores, thr, eo);
  hipLaunchKernelGGL(final_kernel, dim3((N_TOK * C_DIM) / 256), dim3(256), 0, stream,
                     eo, fin);
}
